// Model_39676907882762
// MI455X (gfx1250) — matrix operand order repaired and checked by reading, not compiled, not run on hardware
//
#include <hip/hip_runtime.h>

typedef __attribute__((ext_vector_type(16))) _Float16 v16h;
typedef __attribute__((ext_vector_type(8)))  _Float16 v8h;
typedef __attribute__((ext_vector_type(8)))  float    v8f;
typedef __attribute__((ext_vector_type(2)))  float    v2f;
typedef int vb128_t __attribute__((vector_size(16)));   // matches builtin param type

#define LQ 2048
#define LK 2048
#define DD 128
#define TK 32          // k-tile width
#define NW 4           // waves per block (wave32)

#define AS1 __attribute__((address_space(1)))
#define AS3 __attribute__((address_space(3)))

#if __has_builtin(__builtin_amdgcn_global_load_async_to_lds_b128) && \
    __has_builtin(__builtin_amdgcn_global_load_async_to_lds_b32)
#define HAVE_ASYNC 1
#else
#define HAVE_ASYNC 0
#endif

#if HAVE_ASYNC && __has_builtin(__builtin_amdgcn_s_wait_asynccnt)
#define WAIT_ASYNC0() __builtin_amdgcn_s_wait_asynccnt(0)
#elif HAVE_ASYNC
#define WAIT_ASYNC0() asm volatile("s_wait_asynccnt 0" ::: "memory")
#else
#define WAIT_ASYNC0()
#endif

// ---- cross-lane reductions within each 16-lane half (C/D row layout) ----
#if __has_builtin(__builtin_amdgcn_ds_swizzle)
template <int IMM>
__device__ __forceinline__ float swz_f(float x) {
    return __builtin_bit_cast(float,
        __builtin_amdgcn_ds_swizzle(__builtin_bit_cast(int, x), IMM));
}
__device__ __forceinline__ float half_max16(float x) {
    x = fmaxf(x, swz_f<0x041f>(x));   // xor 1
    x = fmaxf(x, swz_f<0x081f>(x));   // xor 2
    x = fmaxf(x, swz_f<0x101f>(x));   // xor 4
    x = fmaxf(x, swz_f<0x201f>(x));   // xor 8
    return x;
}
__device__ __forceinline__ float half_sum16(float x) {
    x += swz_f<0x041f>(x);
    x += swz_f<0x081f>(x);
    x += swz_f<0x101f>(x);
    x += swz_f<0x201f>(x);
    return x;
}
#else
__device__ __forceinline__ float half_max16(float x) {
#pragma unroll
    for (int off = 1; off < 16; off <<= 1) x = fmaxf(x, __shfl_xor(x, off, 32));
    return x;
}
__device__ __forceinline__ float half_sum16(float x) {
#pragma unroll
    for (int off = 1; off < 16; off <<= 1) x += __shfl_xor(x, off, 32);
    return x;
}
#endif

__global__ __launch_bounds__(NW * 32)
void fattn_sf_drop_kernel(const float* __restrict__ Qg,
                          const float* __restrict__ Kg,
                          const float* __restrict__ Vg,
                          const float* __restrict__ SFg,
                          const float* __restrict__ Ug,
                          float* __restrict__ Og)
{
#if HAVE_ASYNC
    __shared__ __align__(16) float KsF[2][TK * DD];   // [d][kcol] = d*TK + kcol, 16 KB each
    __shared__ __align__(16) float VsT[2][DD * TK];   // [d][l]    = d*TK + l,   16 KB each
#else
    __shared__ __align__(16) float KsF[1][TK * DD];
    __shared__ __align__(16) float VsT[1][DD * TK];
#endif
    __shared__ __align__(16) _Float16 Ps[NW][16 * TK];  // per-wave P tile [m][k]

    const int tid  = threadIdx.x;
    const int wv   = tid >> 5;
    const int lane = tid & 31;
    const int n16  = lane & 15;     // WMMA B/C/D column, A row
    const int hi   = lane >> 4;     // lane half

    const int b  = blockIdx.x >> 5;
    const int q0 = (blockIdx.x & 31) * (NW * 16);
    const int qw = q0 + wv * 16;

    const float* Qp  = Qg  + (size_t)b * LQ * DD;
    const float* Kp  = Kg  + (size_t)b * DD * LK;
    const float* Vp  = Vg  + (size_t)b * LK * DD;
    const float* SFp = SFg + (size_t)b * LQ * LK;
    const float* Up  = Ug  + (size_t)b * LQ * LK;
    float*       Op  = Og  + (size_t)b * LQ * DD;

    // ---- resident Q fragments: A 16x4 f32, lane row m=n16, K = 4t+2*hi+{0,1} ----
    v2f aq[DD / 4];
    {
        const float* qr = Qp + (size_t)(qw + n16) * DD;
#pragma unroll
        for (int t = 0; t < DD / 4; ++t) {
            aq[t][0] = qr[4 * t + 2 * hi + 0];
            aq[t][1] = qr[4 * t + 2 * hi + 1];
        }
    }

    v8f o[8];
    const v8f vzero = {0.f, 0.f, 0.f, 0.f, 0.f, 0.f, 0.f, 0.f};
#pragma unroll
    for (int c = 0; c < 8; ++c) o[c] = vzero;
    float mrow[8], lrow[8];
#pragma unroll
    for (int r = 0; r < 8; ++r) { mrow[r] = -3.0e38f; lrow[r] = 0.0f; }

#if HAVE_ASYNC
    // ---- async DMA of one K/V tile into LDS buffer bn (no VGPR staging) ----
    auto issue_copy = [&](int bn, int kt) {
#pragma unroll
        for (int i = 0; i < 8; ++i) {            // K: 32x128 f32, B128 per lane
            int idx = tid + 128 * i;             // 0..1023
            int d   = idx >> 3;                  // 0..127
            int kq  = (idx & 7) * 4;             // 0..28
            AS1 vb128_t* g = (AS1 vb128_t*)(size_t)(Kp + (size_t)d * LK + kt + kq);
            AS3 vb128_t* l = (AS3 vb128_t*)(unsigned int)(size_t)&KsF[bn][d * TK + kq];
            __builtin_amdgcn_global_load_async_to_lds_b128(g, l, 0, 0);
        }
#pragma unroll
        for (int i = 0; i < 32; ++i) {           // V: transpose at 4B granularity
            int e  = tid + 128 * i;              // 0..4095
            int d  = e & 127;
            int lr = e >> 7;                     // 0..31
            AS1 int* g = (AS1 int*)(size_t)(Vp + (size_t)(kt + lr) * DD + d);
            AS3 int* l = (AS3 int*)(unsigned int)(size_t)&VsT[bn][d * TK + lr];
            __builtin_amdgcn_global_load_async_to_lds_b32(g, l, 0, 0);
        }
    };
#else
    // ---- fallback: batched register-staged copy (loads first, then stores) ----
    auto stage_copy = [&](int kt) {
        float4 tk[8], tv[8];
#pragma unroll
        for (int i = 0; i < 8; ++i) {
            int idx = tid + 128 * i; int d = idx >> 3; int kq = (idx & 7) * 4;
            tk[i] = *(const float4*)(Kp + (size_t)d * LK + kt + kq);
        }
#pragma unroll
        for (int i = 0; i < 8; ++i) {
            int idx = tid + 128 * i; int lr = idx >> 5; int dq = (idx & 31) * 4;
            tv[i] = *(const float4*)(Vp + (size_t)(kt + lr) * DD + dq);
        }
#pragma unroll
        for (int i = 0; i < 8; ++i) {
            int idx = tid + 128 * i; int d = idx >> 3; int kq = (idx & 7) * 4;
            *(float4*)&KsF[0][d * TK + kq] = tk[i];
        }
#pragma unroll
        for (int i = 0; i < 8; ++i) {
            int idx = tid + 128 * i; int lr = idx >> 5; int dq = (idx & 31) * 4;
            VsT[0][(dq + 0) * TK + lr] = tv[i].x;
            VsT[0][(dq + 1) * TK + lr] = tv[i].y;
            VsT[0][(dq + 2) * TK + lr] = tv[i].z;
            VsT[0][(dq + 3) * TK + lr] = tv[i].w;
        }
    };
#endif

    // ---- one k-tile: QK (fp32 WMMA), online softmax + dropout, PV (f16 WMMA) ----
    auto compute_tile = [&](int bn, int kt) {
        const float* Kt = &KsF[bn][0];
        const float* Vt = &VsT[bn][0];
        v8f s0 = vzero, s1 = vzero;
#pragma unroll
        for (int t = 0; t < DD / 4; ++t) {
            v2f b0, b1;   // B 4x16 f32: lane col n, K = 4t+2*hi+{0,1}
            b0[0] = Kt[(4 * t + 2 * hi + 0) * TK + n16];
            b0[1] = Kt[(4 * t + 2 * hi + 1) * TK + n16];
            b1[0] = Kt[(4 * t + 2 * hi + 0) * TK + 16 + n16];
            b1[1] = Kt[(4 * t + 2 * hi + 1) * TK + 16 + n16];
            s0 = __builtin_amdgcn_wmma_f32_16x16x4_f32(false, aq[t], false, b0, (short)0, s0, false, false);
            s1 = __builtin_amdgcn_wmma_f32_16x16x4_f32(false, aq[t], false, b1, (short)0, s1, false, false);
        }

#pragma unroll
        for (int r = 0; r < 8; ++r) {
            const size_t rowoff = (size_t)(qw + 8 * hi + r) * LK + kt;   // VGPR r -> M = r+8*hi
            float x0 = s0[r] * __builtin_nontemporal_load(SFp + rowoff + n16);
            float x1 = s1[r] * __builtin_nontemporal_load(SFp + rowoff + 16 + n16);
            if (kt + TK < LK) {
                __builtin_prefetch(SFp + rowoff + TK, 0, 0);
                __builtin_prefetch(Up  + rowoff + TK, 0, 0);
            }
            float mx    = half_max16(fmaxf(x0, x1));
            float mnew  = fmaxf(mrow[r], mx);
            float alpha = __expf(mrow[r] - mnew);
            mrow[r] = mnew;
            float p0 = __expf(x0 - mnew);
            float p1 = __expf(x1 - mnew);
            // per-lane partial row-sum (same alpha chain in all 16 lanes);
            // cross-lane reduction deferred to the epilogue
            lrow[r] = lrow[r] * alpha + (p0 + p1);
            float u0 = __builtin_nontemporal_load(Up + rowoff + n16);
            float u1 = __builtin_nontemporal_load(Up + rowoff + 16 + n16);
            p0 = (u0 >= 0.5f) ? p0 * 2.0f : 0.0f;   // inverted dropout, p=0.5
            p1 = (u1 >= 0.5f) ? p1 * 2.0f : 0.0f;
            Ps[wv][(8 * hi + r) * TK + n16]      = (_Float16)p0;
            Ps[wv][(8 * hi + r) * TK + 16 + n16] = (_Float16)p1;
#pragma unroll
            for (int c = 0; c < 8; ++c) o[c][r] *= alpha;
        }

        // C/D -> A layout for P via per-wave LDS
        v8h plo = *(const v8h*)&Ps[wv][n16 * TK + 16 * hi];
        v8h phi = *(const v8h*)&Ps[wv][n16 * TK + 16 * hi + 8];
        v16h ap = __builtin_shufflevector(plo, phi, 0,1,2,3,4,5,6,7,8,9,10,11,12,13,14,15);

#pragma unroll
        for (int c = 0; c < 8; ++c) {
            // B 32x16 f16: lane col n = 16c+n16, half j -> K = 16*hi + j
            const float* vp = &Vt[(16 * c + n16) * TK + 16 * hi];
            float4 f0 = *(const float4*)(vp);
            float4 f1 = *(const float4*)(vp + 4);
            float4 f2 = *(const float4*)(vp + 8);
            float4 f3 = *(const float4*)(vp + 12);
            v16h bv;
            bv[0]  = (_Float16)f0.x; bv[1]  = (_Float16)f0.y; bv[2]  = (_Float16)f0.z; bv[3]  = (_Float16)f0.w;
            bv[4]  = (_Float16)f1.x; bv[5]  = (_Float16)f1.y; bv[6]  = (_Float16)f1.z; bv[7]  = (_Float16)f1.w;
            bv[8]  = (_Float16)f2.x; bv[9]  = (_Float16)f2.y; bv[10] = (_Float16)f2.z; bv[11] = (_Float16)f2.w;
            bv[12] = (_Float16)f3.x; bv[13] = (_Float16)f3.y; bv[14] = (_Float16)f3.z; bv[15] = (_Float16)f3.w;
            o[c] = __builtin_amdgcn_wmma_f32_16x16x32_f16(false, ap, false, bv, (short)0, o[c], false, false);
        }
    };

#if HAVE_ASYNC
    // ---- software pipeline: DMA tile i+1 while computing tile i ----
    issue_copy(0, 0);
    WAIT_ASYNC0();
    __syncthreads();
    int buf = 0;
    for (int kk = 0; kk < LK; kk += TK) {
        if (kk + TK < LK) issue_copy(buf ^ 1, kk + TK);
        compute_tile(buf, kk);
        WAIT_ASYNC0();          // this wave's DMA for tile i+1 done
        __syncthreads();        // everyone's DMA done, tile i reads done
        buf ^= 1;
    }
#else
    for (int kk = 0; kk < LK; kk += TK) {
        __syncthreads();
        stage_copy(kk);
        __syncthreads();
        compute_tile(0, kk);
    }
#endif

    // ---- epilogue: finish deferred row-sum reduction, normalize, store ----
#pragma unroll
    for (int r = 0; r < 8; ++r) {
        float rl = 1.0f / half_sum16(lrow[r]);
        const size_t rowoff = (size_t)(qw + 8 * hi + r) * DD;
#pragma unroll
        for (int c = 0; c < 8; ++c) {
            __builtin_nontemporal_store(o[c][r] * rl, Op + rowoff + 16 * c + n16);
        }
    }
}

extern "C" void kernel_launch(void* const* d_in, const int* in_sizes, int n_in,
                              void* d_out, int out_size, void* d_ws, size_t ws_size,
                              hipStream_t stream) {
    const float* Q  = (const float*)d_in[0];   // [8,2048,128]
    const float* K  = (const float*)d_in[1];   // [8,128,2048]
    const float* V  = (const float*)d_in[2];   // [8,2048,128]
    const float* SF = (const float*)d_in[3];   // [8,2048,2048]
    const float* U  = (const float*)d_in[4];   // [8,2048,2048]
    float* O = (float*)d_out;                  // [8,2048,128]

    dim3 grid(8 * (LQ / (NW * 16)));           // 256 blocks
    dim3 block(NW * 32);                       // 128 threads = 4 wave32
    fattn_sf_drop_kernel<<<grid, block, 0, stream>>>(Q, K, V, SF, U, O);
}
